// WindowAttentionV2_77017353552085
// MI455X (gfx1250) — matrix operand order repaired and checked by reading, not compiled, not run on hardware
//
#include <hip/hip_runtime.h>
#include <hip/hip_bf16.h>
#include <math.h>

typedef __attribute__((ext_vector_type(16))) _Float16 v16h;
typedef __attribute__((ext_vector_type(8)))  float    v8f;
typedef __attribute__((ext_vector_type(4)))  unsigned int u32x4;
typedef __attribute__((ext_vector_type(8)))  int      i32x8;
typedef __attribute__((ext_vector_type(4)))  int      i32x4;

#define WMMA(a, b, c) __builtin_amdgcn_wmma_f32_16x16x32_f16( \
    false, (a), false, (b), (short)0, (c), false, false)

#define HAS_TDM __has_builtin(__builtin_amdgcn_tensor_load_to_lds)

// ---------------- WMMA fragment loaders (gfx1250 wave32 layouts) ------------
// A 16x32 f16, row-major src, row stride ld.
// lanes 0-15: row m=lane, K={0..15}; lanes 16-31: row m=lane-16, K={16..31}
__device__ __forceinline__ v16h ld_a_lds(const _Float16* __restrict__ base, int ld, int lane) {
  const int m  = lane & 15;
  const int kb = (lane & 16) ? 16 : 0;
  const _Float16* r = base + m * ld;
  v16h a;
#pragma unroll
  for (int v = 0; v < 8; ++v) {
    const int k = kb + 2 * v;
    a[2 * v]     = r[k];
    a[2 * v + 1] = r[k + 1];
  }
  return a;
}

// B 32x16 f16 where memory holds B[k][n] = base[k*ld + n]
__device__ __forceinline__ v16h ld_b_lds(const _Float16* __restrict__ base, int ld, int lane) {
  const int n  = lane & 15;
  const int kb = (lane & 16) ? 16 : 0;
  v16h b;
#pragma unroll
  for (int v = 0; v < 8; ++v) {
    b[2 * v]     = base[(kb + 2 * v) * ld + n];
    b[2 * v + 1] = base[(kb + 2 * v + 1) * ld + n];
  }
  return b;
}

// B 32x16 f16 where memory holds B[k][n] = base[n*ld + k]  (K^T from row-major K)
__device__ __forceinline__ v16h ld_bT_lds(const _Float16* __restrict__ base, int ld, int lane) {
  const int n  = lane & 15;
  const int kb = (lane & 16) ? 16 : 0;
  const _Float16* r = base + n * ld;
  v16h b;
#pragma unroll
  for (int v = 0; v < 8; ++v) {
    b[2 * v]     = r[kb + 2 * v];
    b[2 * v + 1] = r[kb + 2 * v + 1];
  }
  return b;
}

// B 32x16 from pre-converted f16 weight (row-major [out][in]); B[k][n]=W[n0+n][k0+k]
// per-lane: 16 contiguous halves, 32B aligned -> 2x global_load_b128
__device__ __forceinline__ v16h ld_b_wgt(const _Float16* __restrict__ W, int ldw,
                                         int n0, int k0, int lane) {
  const _Float16* p = W + (size_t)(n0 + (lane & 15)) * ldw + k0 + ((lane & 16) ? 16 : 0);
  return *(const v16h*)p;
}

// -------------------- weight pre-conversion (fp32 -> f16, once) -------------
__global__ void swinv2_cvt_w_kernel(const float* __restrict__ qkv_w,
                                    const float* __restrict__ proj_w,
                                    _Float16* __restrict__ qkv_wh,
                                    _Float16* __restrict__ proj_wh) {
  const int t = blockIdx.x * blockDim.x + threadIdx.x;
  const int stride = blockDim.x * gridDim.x;
  for (int i = t; i < 768 * 256; i += stride) qkv_wh[i] = (_Float16)qkv_w[i];
  for (int i = t; i < 256 * 256; i += stride) proj_wh[i] = (_Float16)proj_w[i];
}

// -------------------- tiny MLP relative-position-bias kernel ----------------
__global__ void swinv2_bias_kernel(const float* __restrict__ w1, const float* __restrict__ b1,
                                   const float* __restrict__ w2, const float* __restrict__ b2,
                                   float* __restrict__ bias) {
  const int t = blockIdx.x * blockDim.x + threadIdx.x;
  for (int p = t; p < 64 * 64; p += blockDim.x * gridDim.x) {
    const int i = p >> 6, j = p & 63;
    const float d0 = (float)((i >> 3) - (j >> 3));
    const float d1 = (float)((i & 7) - (j & 7));
    const float s0 = (d0 > 0.f) ? 1.f : ((d0 < 0.f) ? -1.f : 0.f);
    const float s1 = (d1 > 0.f) ? 1.f : ((d1 < 0.f) ? -1.f : 0.f);
    const float r0 = s0 * log1pf(fabsf(d0));
    const float r1 = s1 * log1pf(fabsf(d1));
    float acc[8];
#pragma unroll
    for (int h = 0; h < 8; ++h) acc[h] = 0.f;
    for (int hh = 0; hh < 256; ++hh) {
      const float hv = fmaxf(r0 * w1[hh * 2] + r1 * w1[hh * 2 + 1] + b1[hh], 0.f);
#pragma unroll
      for (int h = 0; h < 8; ++h) acc[h] += hv * w2[h * 256 + hh];
    }
#pragma unroll
    for (int h = 0; h < 8; ++h) bias[h * 4096 + p] = acc[h] + b2[h];
  }
}

// -------------------- fused window-attention kernel -------------------------
__launch_bounds__(256, 1)
__global__ void swinv2_attn_kernel(const float* __restrict__ x,
                                   const float* __restrict__ mask, int nW,
                                   const _Float16* __restrict__ qkv_wh,
                                   const float* __restrict__ qkv_b,
                                   const _Float16* __restrict__ proj_wh,
                                   const float* __restrict__ proj_b,
                                   const float* __restrict__ tau,
                                   const float* __restrict__ bias,
                                   float* __restrict__ out) {
  __shared__ _Float16 sX[64 * 256];          // x (f16); reused as proj input
  __shared__ _Float16 sQKV[3 * 8 * 64 * 32]; // [which][head][row][c] f16
  __shared__ float    sQn[8][64];
  __shared__ float    sKn[8][64];
  __shared__ _Float16 sA[8][64 * 72];        // per-head softmaxed attn, padded
  __shared__ float    sMask[64 * 64];        // window mask (TDM target)

  const int b    = blockIdx.x;
  const int tid  = threadIdx.x;
  const int wave = __builtin_amdgcn_readfirstlane(tid >> 5);  // scalar wave id
  const int lane = tid & 31;
  const int cb   = lane & 15;                // C/D column within tile
  const int ro   = (lane >> 4) << 3;         // C/D row offset
  const float* __restrict__ xb    = x + (size_t)b * 64 * 256;
  const float* __restrict__ maskg = mask + (size_t)(b % nW) * 4096;

#if HAS_TDM
  // ---- kick off the window-mask DMA (overlaps with QKV GEMM) ----
  if (tid == 0) {
    const unsigned long long ga = (unsigned long long)(const void*)maskg;
    const unsigned lds_off = (unsigned)(size_t)(void*)&sMask[0];
    const u32x4 g0 = { 1u,                                     // count=1 user D#
                       lds_off,                                // lds_addr
                       (unsigned)(ga & 0xffffffffu),
                       (unsigned)((ga >> 32) & 0x01ffffffu) | (2u << 30) }; // type=2
    const i32x8 g1 = { (int)(2u << 16),        // data_size = 4 bytes
                       (int)(4096u << 16),     // tensor_dim0 = 4096 (lo16 @ bits63:48)
                       (int)(1u << 16),        // dim0 hi=0 | tensor_dim1 = 1
                       (int)(4096u << 16),     // dim1 hi=0 | tile_dim0 = 4096
                       1,                      // tile_dim1 = 1, tile_dim2 = 0
                       4096,                   // tensor_dim0_stride (lo32)
                       0, 0 };
    const i32x4 gz = { 0, 0, 0, 0 };
#if defined(__clang_major__) && __clang_major__ >= 23
    const i32x8 g4 = { 0, 0, 0, 0, 0, 0, 0, 0 };
    __builtin_amdgcn_tensor_load_to_lds(g0, g1, gz, gz, g4, 0);
#else
    __builtin_amdgcn_tensor_load_to_lds(g0, g1, gz, gz, 0);
#endif
  }
#endif

  if (wave == 0) {
    __builtin_prefetch(qkv_wh + (size_t)lane * 4096, 0, 3);
    __builtin_prefetch(proj_wh + (size_t)lane * 2048, 0, 3);
  }

  // ---- stage x as f16 ----
  {
    const float4* x4 = (const float4*)xb;
    for (int i = tid; i < 64 * 256 / 4; i += 256) {
      const float4 v = x4[i];
      sX[4 * i + 0] = (_Float16)v.x;
      sX[4 * i + 1] = (_Float16)v.y;
      sX[4 * i + 2] = (_Float16)v.z;
      sX[4 * i + 3] = (_Float16)v.w;
    }
#if !HAS_TDM
    for (int i = tid; i < 4096; i += 256) sMask[i] = maskg[i];
#endif
  }
  __syncthreads();

  // ---- QKV GEMM: [64x256] @ qkv_w^T -> [64x768]; wave owns 6 N-tiles ----
  for (int ntl = 0; ntl < 6; ++ntl) {
    const int n0 = wave * 96 + ntl * 16;     // scalar
    v8f acc[4];
#pragma unroll
    for (int mt = 0; mt < 4; ++mt) { v8f z = {}; acc[mt] = z; }
    for (int kk = 0; kk < 8; ++kk) {
      const v16h bf = ld_b_wgt(qkv_wh, 256, n0, kk * 32, lane);
#pragma unroll
      for (int mt = 0; mt < 4; ++mt) {
        const v16h af = ld_a_lds(sX + mt * 16 * 256 + kk * 32, 256, lane);
        acc[mt] = WMMA(af, bf, acc[mt]);
      }
    }
    // epilogue: +bias, f16 into unified per-head LDS (pure address arithmetic)
    _Float16* dst = sQKV + (n0 >> 8) * (8 * 64 * 32)   // which: q/k/v
                  + ((n0 & 255) >> 5) * (64 * 32);     // head
    const int c0 = n0 & 31;                            // 0 or 16 within head
    const float bv = qkv_b[n0 + cb];
#pragma unroll
    for (int mt = 0; mt < 4; ++mt) {
#pragma unroll
      for (int r = 0; r < 8; ++r) {
        const int row = mt * 16 + r + ro;
        dst[row * 32 + c0 + cb] = (_Float16)(acc[mt][r] + bv);
      }
    }
  }
#if HAS_TDM
  if (tid < 32) __builtin_amdgcn_s_wait_tensorcnt(0);
#endif
  __syncthreads();

  // ---- per-head cosine attention (wave = head) ----
  {
    const int h = wave;
    const _Float16* __restrict__ sQh = sQKV + 0 * (8 * 64 * 32) + h * (64 * 32);
    const _Float16* __restrict__ sKh = sQKV + 1 * (8 * 64 * 32) + h * (64 * 32);
    const _Float16* __restrict__ sVh = sQKV + 2 * (8 * 64 * 32) + h * (64 * 32);
    const float inv_tau = 1.0f / fmaxf(tau[h], 0.01f);
    const float* __restrict__ biash = bias + (size_t)h * 4096;

    // row norms (f16 source, f32 accumulate) — vectorizes to ds_load_b128
    for (int row = lane; row < 64; row += 32) {
      const _Float16* qr = sQh + row * 32;
      const _Float16* kr = sKh + row * 32;
      float sq = 0.f, sk = 0.f;
#pragma unroll
      for (int c = 0; c < 32; ++c) {
        const float qv = (float)qr[c], kv = (float)kr[c];
        sq += qv * qv;
        sk += kv * kv;
      }
      sQn[h][row] = sqrtf(sq);
      sKn[h][row] = sqrtf(sk);
    }

    // Q K^T : K-dim = 32 -> single WMMA step per tile
    v8f att[4][4];
#pragma unroll
    for (int mt = 0; mt < 4; ++mt) {
      const v16h qf = ld_a_lds(sQh + mt * 16 * 32, 32, lane);
#pragma unroll
      for (int nt = 0; nt < 4; ++nt) {
        const v16h kf = ld_bT_lds(sKh + nt * 16 * 32, 32, lane);
        v8f z = {};
        att[mt][nt] = WMMA(qf, kf, z);
      }
    }

    // scale + bias + mask + register-resident softmax (shfl over 16-lane halves)
#pragma unroll
    for (int mt = 0; mt < 4; ++mt) {
#pragma unroll
      for (int r = 0; r < 8; ++r) {
        const int row = mt * 16 + r + ro;
        const float qn = sQn[h][row];
        float m = -3.0e38f;
#pragma unroll
        for (int nt = 0; nt < 4; ++nt) {
          const int col = nt * 16 + cb;
          const float kn = sKn[h][col];
          float a = att[mt][nt][r] / fmaxf(qn * kn, 1e-6f);
          a = a * inv_tau + biash[row * 64 + col] + sMask[row * 64 + col];
          att[mt][nt][r] = a;
          m = fmaxf(m, a);
        }
#pragma unroll
        for (int s = 1; s < 16; s <<= 1) m = fmaxf(m, __shfl_xor(m, s, 32));
        float sum = 0.f;
#pragma unroll
        for (int nt = 0; nt < 4; ++nt) {
          const float e = __expf(att[mt][nt][r] - m);
          att[mt][nt][r] = e;
          sum += e;
        }
#pragma unroll
        for (int s = 1; s < 16; s <<= 1) sum += __shfl_xor(sum, s, 32);
        const float inv = 1.0f / sum;
#pragma unroll
        for (int nt = 0; nt < 4; ++nt)
          sA[h][row * 72 + nt * 16 + cb] = (_Float16)(att[mt][nt][r] * inv);
      }
    }

    // P V : [64x64] @ [64x32]; write head slice into sX (proj input)
    v8f oacc[4][2];
#pragma unroll
    for (int mt = 0; mt < 4; ++mt)
#pragma unroll
      for (int nt = 0; nt < 2; ++nt) { v8f z = {}; oacc[mt][nt] = z; }
    for (int kk = 0; kk < 2; ++kk) {
      v16h vf[2];
#pragma unroll
      for (int nt = 0; nt < 2; ++nt)
        vf[nt] = ld_b_lds(sVh + kk * 32 * 32 + nt * 16, 32, lane);
#pragma unroll
      for (int mt = 0; mt < 4; ++mt) {
        const v16h af = ld_a_lds(&sA[h][mt * 16 * 72 + kk * 32], 72, lane);
#pragma unroll
        for (int nt = 0; nt < 2; ++nt) oacc[mt][nt] = WMMA(af, vf[nt], oacc[mt][nt]);
      }
    }
#pragma unroll
    for (int mt = 0; mt < 4; ++mt)
#pragma unroll
      for (int nt = 0; nt < 2; ++nt)
#pragma unroll
        for (int r = 0; r < 8; ++r) {
          const int row = mt * 16 + r + ro;
          sX[row * 256 + h * 32 + nt * 16 + cb] = (_Float16)oacc[mt][nt][r];
        }
  }
  __syncthreads();

  // ---- output projection: [64x256] @ proj_w^T -> d_out (f32) ----
  float* __restrict__ ob = out + (size_t)b * 64 * 256;
  for (int ntl = 0; ntl < 2; ++ntl) {
    const int n0 = wave * 32 + ntl * 16;
    v8f acc[4];
#pragma unroll
    for (int mt = 0; mt < 4; ++mt) { v8f z = {}; acc[mt] = z; }
    for (int kk = 0; kk < 8; ++kk) {
      const v16h bf = ld_b_wgt(proj_wh, 256, n0, kk * 32, lane);
#pragma unroll
      for (int mt = 0; mt < 4; ++mt) {
        const v16h af = ld_a_lds(sX + mt * 16 * 256 + kk * 32, 256, lane);
        acc[mt] = WMMA(af, bf, acc[mt]);
      }
    }
    const float pb = proj_b[n0 + cb];
#pragma unroll
    for (int mt = 0; mt < 4; ++mt)
#pragma unroll
      for (int r = 0; r < 8; ++r) {
        const int row = mt * 16 + r + ro;
        ob[row * 256 + n0 + cb] = acc[mt][r] + pb;
      }
  }
}

extern "C" void kernel_launch(void* const* d_in, const int* in_sizes, int n_in,
                              void* d_out, int out_size, void* d_ws, size_t ws_size,
                              hipStream_t stream) {
  const float* x      = (const float*)d_in[0];
  const float* mask   = (const float*)d_in[1];
  const float* qkv_w  = (const float*)d_in[2];
  const float* qkv_b  = (const float*)d_in[3];
  const float* proj_w = (const float*)d_in[4];
  const float* proj_b = (const float*)d_in[5];
  const float* m_w1   = (const float*)d_in[6];
  const float* m_b1   = (const float*)d_in[7];
  const float* m_w2   = (const float*)d_in[8];
  const float* m_b2   = (const float*)d_in[9];
  const float* tau    = (const float*)d_in[10];
  float* out = (float*)d_out;

  // workspace layout: bias f32 | qkv_w f16 | proj_w f16   (~640 KB total)
  char* ws = (char*)d_ws;
  float*     bias    = (float*)ws;                        // 32768 f32
  _Float16*  qkv_wh  = (_Float16*)(ws + 131072);          // 196608 f16
  _Float16*  proj_wh = (_Float16*)(ws + 131072 + 393216); // 65536 f16

  const int B_ = in_sizes[0] / (64 * 256);                // 2048 windows
  const int nW = in_sizes[1] / (64 * 64);                 // 512 masks

  hipLaunchKernelGGL(swinv2_cvt_w_kernel, dim3(128), dim3(256), 0, stream,
                     qkv_w, proj_w, qkv_wh, proj_wh);
  hipLaunchKernelGGL(swinv2_bias_kernel, dim3(16), dim3(256), 0, stream,
                     m_w1, m_b1, m_w2, m_b2, bias);
  hipLaunchKernelGGL(swinv2_attn_kernel, dim3(B_), dim3(256), 0, stream,
                     x, mask, nW, qkv_wh, qkv_b, proj_wh, proj_b, tau, bias, out);
}
